// BP_FNN_70042326663678
// MI455X (gfx1250) — compile-verified, matrix operand order checked by reading, not run on hardware
//
#include <hip/hip_runtime.h>

// ---------------- types ----------------
typedef __bf16 bf16_t;
typedef __attribute__((ext_vector_type(16))) __bf16 v16bf;
typedef __attribute__((ext_vector_type(8)))  float  v8f;

// ---------------- problem constants ----------------
#define BATCH   8192
#define NRULE   256
#define NFEA    128
#define MBLK    128           // batch rows per workgroup (8 waves x 16 rows)
#define LDA     264           // LDS row stride (bf16 elems), padded: conflict-free b128 frags
#define LDBS    264
#define LDBQ    136
#define RULE_OFFSET (-28.0f)  // the 10^-18 (XOR!) bug constant
#define QCLAMP  1.0e8f

#define SMEM_A   (MBLK  * LDA)   // bf16 elems
#define SMEM_BS  (NRULE * LDBS)
#define SMEM_BQ  (NRULE * LDBQ)
#define SMEM_BYTES ((size_t)(SMEM_A + SMEM_BS + SMEM_BQ) * 2 + (size_t)NRULE * 4 * 2)

// ---------------- WMMA helpers ----------------
__device__ __forceinline__ v8f wmma_bf16(v16bf a, v16bf b, v8f c) {
  // D = A(16x32 bf16) x B(32x16 bf16) + C(16x16 f32)
  return __builtin_amdgcn_wmma_f32_16x16x32_bf16(
      /*neg_a=*/false, a, /*neg_b=*/false, b,
      /*c_mod=*/(short)0, c, /*reuse_a=*/false, /*reuse_b=*/false);
}

// Load one 16x32 bf16 fragment from an LDS row-major [16][K] tile.
// Per ISA 16-bit A layout: lanes 0-15 hold K {kc*32+0..7, kc*32+16..23},
// lanes 16-31 hold K {kc*32+8..15, kc*32+24..31}; row = lane & 15.
__device__ __forceinline__ v16bf load_frag(const bf16_t* rowptr, int kc, int laneHi) {
  const bf16_t* p = rowptr + kc * 32 + laneHi * 8;
  union { uint4 q[2]; v16bf v; } u;
  u.q[0] = *(const uint4*)(p);
  u.q[1] = *(const uint4*)(p + 16);
  return u.v;
}

// ---------------- fused TSK-FNN forward ----------------
__global__ __launch_bounds__(256, 1)
void tsk_fnn_fused(const float* __restrict__ data,
                   const float* __restrict__ para_mu,
                   const float* __restrict__ para_sigma,
                   const float* __restrict__ para_w3,
                   float* __restrict__ out) {
  extern __shared__ char smem[];
  bf16_t* sA  = (bf16_t*)smem;          // [128][LDA]  : [x | x^2]  (K = 256)
  bf16_t* sBs = sA  + SMEM_A;           // [256][LDBS] : [2*mu*q | -q]
  bf16_t* sBq = sBs + SMEM_BS;          // [256][LDBQ] : w3[:,1:]
  float*  sDr = (float*)(sBq + SMEM_BQ);// [256]       : -sum mu^2*q
  float*  sW0 = sDr + NRULE;            // [256]       : w3[:,0]

  const int tid = threadIdx.x;
  const int m0  = blockIdx.x * MBLK;

  // ---- stage A = [x ; x^2] as bf16 (vectorized f32x4 global reads) ----
  for (int v = tid; v < MBLK * NFEA / 4; v += 256) {
    const int row = v >> 5;             // 32 float4 per row
    const int c4  = (v & 31) << 2;
    const float4 x = *(const float4*)(data + (size_t)(m0 + row) * NFEA + c4);
    union { bf16_t h[4]; uint2 u; } lin, sq;
    lin.h[0] = (bf16_t)x.x;            sq.h[0] = (bf16_t)(x.x * x.x);
    lin.h[1] = (bf16_t)x.y;            sq.h[1] = (bf16_t)(x.y * x.y);
    lin.h[2] = (bf16_t)x.z;            sq.h[2] = (bf16_t)(x.z * x.z);
    lin.h[3] = (bf16_t)x.w;            sq.h[3] = (bf16_t)(x.w * x.w);
    *(uint2*)(sA + row * LDA + c4)        = lin.u;
    *(uint2*)(sA + row * LDA + NFEA + c4) = sq.u;
  }

  // ---- stage B_S = [2*mu*q ; -q], q = min(1/(2 sigma^2), QCLAMP) ----
  for (int v = tid; v < NRULE * NFEA / 4; v += 256) {
    const int r  = v >> 5;
    const int c4 = (v & 31) << 2;
    const float4 m4 = *(const float4*)(para_mu    + (size_t)r * NFEA + c4);
    const float4 s4 = *(const float4*)(para_sigma + (size_t)r * NFEA + c4);
    const float mv[4] = {m4.x, m4.y, m4.z, m4.w};
    const float sv[4] = {s4.x, s4.y, s4.z, s4.w};
    union { bf16_t h[4]; uint2 u; } bc, ba;
#pragma unroll
    for (int j = 0; j < 4; ++j) {
      const float q = fminf(0.5f / (sv[j] * sv[j]), QCLAMP);
      bc.h[j] = (bf16_t)(2.0f * mv[j] * q);
      ba.h[j] = (bf16_t)(-q);
    }
    *(uint2*)(sBs + r * LDBS + c4)        = bc.u;
    *(uint2*)(sBs + r * LDBS + NFEA + c4) = ba.u;
  }

  // ---- stage B_Q = w3[:,1:] (stride-129 source -> scalar loads) ----
  for (int v = tid; v < NRULE * NFEA / 4; v += 256) {
    const int r  = v >> 5;
    const int c4 = (v & 31) << 2;
    const float* wr = para_w3 + (size_t)r * (NFEA + 1) + 1 + c4;
    union { bf16_t h[4]; uint2 u; } bw;
    bw.h[0] = (bf16_t)wr[0]; bw.h[1] = (bf16_t)wr[1];
    bw.h[2] = (bf16_t)wr[2]; bw.h[3] = (bf16_t)wr[3];
    *(uint2*)(sBq + r * LDBQ + c4) = bw.u;
  }

  // ---- per-rule constants: D_r (f32, float4-vectorized) and w0 ----
  {
    const int r = tid;                  // blockDim.x == NRULE == 256
    const float4* mr4 = (const float4*)(para_mu    + (size_t)r * NFEA);
    const float4* sr4 = (const float4*)(para_sigma + (size_t)r * NFEA);
    float dsum = 0.0f;
#pragma unroll 4
    for (int f4 = 0; f4 < NFEA / 4; ++f4) {
      const float4 m = mr4[f4];
      const float4 s = sr4[f4];
      dsum -= m.x * m.x * fminf(0.5f / (s.x * s.x), QCLAMP);
      dsum -= m.y * m.y * fminf(0.5f / (s.y * s.y), QCLAMP);
      dsum -= m.z * m.z * fminf(0.5f / (s.z * s.z), QCLAMP);
      dsum -= m.w * m.w * fminf(0.5f / (s.w * s.w), QCLAMP);
    }
    sDr[r] = dsum;
    sW0[r] = para_w3[(size_t)r * (NFEA + 1)];
  }
  __syncthreads();

  // ---- WMMA main loop: each wave owns 16 batch rows x all 256 rules ----
  const int lane   = tid & 31;
  const int wave   = tid >> 5;
  const int laneLo = lane & 15;
  const int laneHi = lane >> 4;
  const bf16_t* aRow = sA + (wave * 16 + laneLo) * LDA;

  // A fragments: loaded ONCE per wave, live across the whole rule loop (64 VGPRs).
  v16bf afr[8];
#pragma unroll
  for (int kc = 0; kc < 8; ++kc) afr[kc] = load_frag(aRow, kc, laneHi);

  v8f num = {0.f,0.f,0.f,0.f,0.f,0.f,0.f,0.f};
  v8f den = {0.f,0.f,0.f,0.f,0.f,0.f,0.f,0.f};

  for (int nt = 0; nt < 16; ++nt) {
    const int n0 = nt * 16;
    const bf16_t* bsRow = sBs + (n0 + laneLo) * LDBS;
    const bf16_t* bqRow = sBq + (n0 + laneLo) * LDBQ;

    // Issue ALL B-fragment loads for this tile, in consumption order, BEFORE
    // any WMMA: 24 ds_load_b128 in flight, so each WMMA waits on a
    // progressively smaller DScnt instead of a full LDS round trip.
    const float drv = sDr[n0 + laneLo];
    const float w0v = sW0[n0 + laneLo];
    v16bf bs[8], bq[4];
#pragma unroll
    for (int kc = 0; kc < 4; ++kc) {
      bs[kc] = load_frag(bsRow, kc, laneHi);
      bq[kc] = load_frag(bqRow, kc, laneHi);
    }
#pragma unroll
    for (int kc = 4; kc < 8; ++kc) bs[kc] = load_frag(bsRow, kc, laneHi);

    // Hard fence: scheduler must not sink the loads below this point.
    __builtin_amdgcn_sched_barrier(0);

    v8f accS = {0.f,0.f,0.f,0.f,0.f,0.f,0.f,0.f};
    v8f accQ = {0.f,0.f,0.f,0.f,0.f,0.f,0.f,0.f};
    // Two independent accumulation chains interleaved for ILP.
#pragma unroll
    for (int kc = 0; kc < 4; ++kc) {
      accS = wmma_bf16(afr[kc], bs[kc], accS);
      accQ = wmma_bf16(afr[kc], bq[kc], accQ);
    }
#pragma unroll
    for (int kc = 4; kc < 8; ++kc) {
      accS = wmma_bf16(afr[kc], bs[kc], accS);
    }

    // fused epilogue: rule activation + consequent, accumulate num/den
    // (next iteration's ds_loads may overlap the WMMA->VALU hazard gap)
#pragma unroll
    for (int i = 0; i < 8; ++i) {
      const float s    = fminf(accS[i] + drv, 0.0f);   // true S <= 0: kill bf16 blow-up
      const float rule = __expf(s) + RULE_OFFSET;
      const float conq = accQ[i] + w0v;
      num[i] += rule * conq;
      den[i] += rule;
    }
  }

  // ---- reduce over rule columns (16 lanes per half-wave), sigmoid, store ----
#pragma unroll
  for (int i = 0; i < 8; ++i) {
    float n = num[i], d = den[i];
    n += __shfl_xor(n, 1, 32);  d += __shfl_xor(d, 1, 32);
    n += __shfl_xor(n, 2, 32);  d += __shfl_xor(d, 2, 32);
    n += __shfl_xor(n, 4, 32);  d += __shfl_xor(d, 4, 32);
    n += __shfl_xor(n, 8, 32);  d += __shfl_xor(d, 8, 32);
    if (laneLo == 0) {
      const float z = n / d;                     // (sum rule*conq)/(sum rule)
      out[m0 + wave * 16 + laneHi * 8 + i] = 1.0f / (1.0f + __expf(-z));
    }
  }
}

// ---------------- host launch ----------------
extern "C" void kernel_launch(void* const* d_in, const int* in_sizes, int n_in,
                              void* d_out, int out_size, void* d_ws, size_t ws_size,
                              hipStream_t stream) {
  (void)in_sizes; (void)n_in; (void)out_size; (void)d_ws; (void)ws_size;
  const float* data  = (const float*)d_in[0];
  const float* mu    = (const float*)d_in[1];
  const float* sigma = (const float*)d_in[2];
  const float* w3    = (const float*)d_in[3];
  float* out = (float*)d_out;

  // 268 KB dynamic LDS (< 320 KB/WGP on CDNA5); raise the dyn-smem cap.
  (void)hipFuncSetAttribute((const void*)tsk_fnn_fused,
                            hipFuncAttributeMaxDynamicSharedMemorySize,
                            (int)SMEM_BYTES);
  tsk_fnn_fused<<<BATCH / MBLK, 256, SMEM_BYTES, stream>>>(data, mu, sigma, w3, out);
}
